// DWM_63299228008955
// MI455X (gfx1250) — matrix operand order repaired and checked by reading, not compiled, not run on hardware
//
#include <hip/hip_runtime.h>
#include <hip/hip_bf16.h>

// ---------------------------------------------------------------------------
// DWM forward for MI455X (gfx1250): persistent cooperative kernel.
//   per step: fused bf16 WMMA GEMM [32x672]x[672x944]  (Ws|Wo|Wu fused, padded)
//             -> sigmoid state / output logits / interface params
//             then memory/addressing update in f32 VALU.
// Weights stay resident in the 192MB L2 as packed bf16 (1.27MB).
// Grid sync: agent-scope atomics (3 barriers per step), s_sleep in spin.
// ---------------------------------------------------------------------------

#define NWG      8
#define BLK      256
#define T_STEPS  512
#define KPAD     672              // 642 -> 21 * 32
#define NPAD     944              // 936 -> 59 * 16
#define NT_TILES 59
#define MT_TILES 2
#define NTILES   (NT_TILES * MT_TILES)   // 118
#define KSTEPS   21
#define WPK_ELEMS (42 * NPAD * 16)       // 42 K-groups-of-16 x 944 cols x 16 halves

typedef __attribute__((ext_vector_type(16))) __bf16 v16bf;
typedef __attribute__((ext_vector_type(8)))  float  v8f;

union FragBF { v16bf v; uint4 u[2]; };

__device__ __forceinline__ unsigned short f2bf(float f) {
  unsigned u = __float_as_uint(f);
  unsigned r = u + 0x7FFFu + ((u >> 16) & 1u);     // RNE
  return (unsigned short)(r >> 16);
}
__device__ __forceinline__ float sigmoidf_(float x) { return 1.f / (1.f + expf(-x)); }
__device__ __forceinline__ float softplusf_(float x) { return (x > 20.f) ? x : log1pf(expf(x)); }

__device__ __forceinline__ void grid_barrier(unsigned* cnt, unsigned* gen) {
  __syncthreads();
  if (threadIdx.x == 0) {
    unsigned g = __hip_atomic_load(gen, __ATOMIC_ACQUIRE, __HIP_MEMORY_SCOPE_AGENT);
    unsigned a = __hip_atomic_fetch_add(cnt, 1u, __ATOMIC_ACQ_REL, __HIP_MEMORY_SCOPE_AGENT);
    if (a + 1u == (unsigned)NWG) {
      __hip_atomic_store(cnt, 0u, __ATOMIC_RELAXED, __HIP_MEMORY_SCOPE_AGENT);
      __hip_atomic_fetch_add(gen, 1u, __ATOMIC_RELEASE, __HIP_MEMORY_SCOPE_AGENT);
    } else {
      while (__hip_atomic_load(gen, __ATOMIC_ACQUIRE, __HIP_MEMORY_SCOPE_AGENT) == g)
        __builtin_amdgcn_s_sleep(1);
    }
  }
  __syncthreads();
}

// ---- pack fused weight matrix to bf16, WMMA-B-friendly K-group-of-16 layout.
// Wpk[ (g*NPAD + n)*16 + e ]  holds element (K = g*16+e, col n); lane's 16
// halves (one K-group) are contiguous = one 32B read.
__global__ void dwm_pack_weights(const float* __restrict__ Ws,
                                 const float* __restrict__ Wo,
                                 const float* __restrict__ Wu,
                                 unsigned short* __restrict__ Wpk) {
  int idx = blockIdx.x * BLK + threadIdx.x;
  if (idx >= WPK_ELEMS) return;
  int e  = idx & 15;
  int nn = (idx >> 4) % NPAD;
  int g  = idx / (NPAD * 16);
  int K  = g * 16 + e;
  float v = 0.f;
  if (K < 642 && nn < 936) {
    if (nn < 256)      v = Ws[K * 256 + nn];
    else if (nn < 512) v = Wo[K * 256 + (nn - 256)];
    else               v = Wu[K * 424 + (nn - 512)];
  }
  Wpk[idx] = f2bf(v);
}

// ---- recurrent-state init ----
__global__ void dwm_init(float* __restrict__ state, float* __restrict__ readv,
                         float* __restrict__ wt, float* __restrict__ wtd,
                         float* __restrict__ mem, unsigned* __restrict__ syncp) {
  int idx = blockIdx.x * blockDim.x + threadIdx.x;
  int stride = gridDim.x * blockDim.x;
  if (idx == 0) { syncp[0] = 0u; syncp[1] = 0u; }
  for (int i = idx; i < 32 * 256; i += stride) state[i] = 1.f;
  for (int i = idx; i < 32 * 128; i += stride) readv[i] = 0.01f;   // wt0 dot mem0
  for (int i = idx; i < 32 * 4 * 512; i += stride) {
    float v = ((i & 511) == 0) ? 1.f : 0.f;
    wt[i] = v; wtd[i] = v;
  }
  for (int i = idx; i < 32 * 32 * 512; i += stride) mem[i] = 0.01f;
}

// ---- persistent step loop ----
__global__ __launch_bounds__(BLK) void dwm_persistent(
    const float* __restrict__ x,
    const float* __restrict__ bs, const float* __restrict__ bo, const float* __restrict__ bu,
    const unsigned short* __restrict__ Wpk,
    unsigned short* __restrict__ comb,
    float* __restrict__ state, float* __restrict__ readv, float* __restrict__ updg,
    float* __restrict__ wt, float* __restrict__ wtd, float* __restrict__ mem,
    float* __restrict__ out, unsigned* __restrict__ syncp) {
  __shared__ float s_wtj[4][512];
  __shared__ float s_z[4][512];
  __shared__ float s_wts[4][512];
  __shared__ float s_norm2[512];
  __shared__ float s_er[4][32], s_addv[4][32], s_kn[4][32];
  __shared__ float s_shift[4][3];
  __shared__ float s_misc[4][8];   // 0:jd 1..3:j 4:gamma 5:beta 6:g 7:|k|
  __shared__ float s_red[8];

  const int tid   = threadIdx.x;
  const int lane  = tid & 31;
  const int waveG = blockIdx.x * (BLK / 32) + (tid >> 5);  // 0..63
  const int hi    = lane >> 4;                             // lane half-select
  const int l16   = lane & 15;
  unsigned* barCnt = syncp;
  unsigned* barGen = syncp + 1;

  for (int t = 0; t < T_STEPS; ++t) {
    // ---- phase 0: build comb = [x_t | state | read] as bf16, grid-coop ----
    for (int idx = blockIdx.x * BLK + tid; idx < 32 * KPAD; idx += NWG * BLK) {
      int bb = idx / KPAD, k = idx % KPAD;
      float v;
      if (k < 258)      v = x[((size_t)bb * 512 + t) * 258 + k];
      else if (k < 514) v = state[bb * 256 + (k - 258)];
      else if (k < 642) v = readv[bb * 128 + (k - 514)];
      else              v = 0.f;
      comb[idx] = f2bf(v);
    }
    grid_barrier(barCnt, barGen);

    // ---- phase 1: fused GEMM via v_wmma_f32_16x16x32_bf16 ----
    for (int tile = waveG; tile < NTILES; tile += NWG * (BLK / 32)) {
      int mT = tile & 1, nT = tile >> 1;
      v8f acc = {};
      const unsigned short* arow = comb + (mT * 16 + l16) * KPAD;
      for (int kk = 0; kk < KSTEPS; ++kk) {
        FragBF a, bm;
        int kbA = kk * 32 + hi * 16;              // A: contiguous K-group-of-16 per lane half
        a.u[0] = *(const uint4*)(arow + kbA);
        a.u[1] = *(const uint4*)(arow + kbA + 8);
        int g = kk * 2 + hi;                      // B: K-group-of-16 per lane half
        const unsigned short* bp = Wpk + ((size_t)g * NPAD + nT * 16 + l16) * 16;
        bm.u[0] = *(const uint4*)(bp);
        bm.u[1] = *(const uint4*)(bp + 8);
        if (kk + 1 < KSTEPS)
          __builtin_prefetch((const void*)(bp + 2 * NPAD * 16), 0, 1);
        acc = __builtin_amdgcn_wmma_f32_16x16x32_bf16(
            false, a.v, false, bm.v, (short)0, acc, false, false);
      }
      int n = nT * 16 + l16;
      if (n < 936) {
        for (int v = 0; v < 8; ++v) {
          int bb = mT * 16 + v + hi * 8;          // C/D: lane/VGPR layout
          float val = acc[v];
          if (n < 256) {
            state[bb * 256 + n] = sigmoidf_(val + bs[n]);
          } else if (n < 512) {
            int c = n - 256;
            out[((size_t)bb * 512 + t) * 256 + c] = val + bo[c];
          } else {
            int c = n - 512;
            updg[bb * 424 + c] = val + bu[c];
          }
        }
      }
    }
    grid_barrier(barCnt, barGen);

    // ---- phase 2: memory + addressing update, batch-per-workgroup ----
    for (int b = blockIdx.x; b < 32; b += NWG) {
      const float* u_base = updg + b * 424;
      if (tid < 4) {                              // per-head scalars
        const float* u = u_base + tid * 106;
        float s0 = softplusf_(u[0]), s1 = softplusf_(u[1]), s2 = softplusf_(u[2]);
        float mx = fmaxf(s0, fmaxf(s1, s2));
        float e0 = expf(s0 - mx), e1 = expf(s1 - mx), e2 = expf(s2 - mx);
        float es = e0 + e1 + e2;
        s_shift[tid][0] = e0 / es; s_shift[tid][1] = e1 / es; s_shift[tid][2] = e2 / es;
        s_misc[tid][0] = sigmoidf_(u[3]);
        float j0 = u[4], j1 = u[5], j2 = u[6];
        float jm = fmaxf(j0, fmaxf(j1, j2));
        float f0 = expf(j0 - jm), f1 = expf(j1 - jm), f2 = expf(j2 - jm);
        float fs = f0 + f1 + f2;
        s_misc[tid][1] = f0 / fs; s_misc[tid][2] = f1 / fs; s_misc[tid][3] = f2 / fs;
        s_misc[tid][4] = 1.f + softplusf_(u[7]);
        s_misc[tid][5] = softplusf_(u[104]);
        s_misc[tid][6] = sigmoidf_(u[105]);
      }
      if (tid < 128) {                            // erase / add / key
        int h = tid >> 5, m = tid & 31;
        const float* u = u_base + h * 106;
        s_er[h][m]   = sigmoidf_(u[8 + m]);
        s_addv[h][m] = u[40 + m];
        s_kn[h][m]   = tanhf(u[72 + m]);
      }
      __syncthreads();
      if (tid < 4) {
        float nrm = 0.f;
        for (int m = 0; m < 32; ++m) nrm += s_kn[tid][m] * s_kn[tid][m];
        s_misc[tid][7] = sqrtf(nrm) + 1e-12f;
      }
      for (int idx = tid; idx < 4 * 512; idx += BLK) {   // wt_d_new, wt_j
        int h = idx >> 9, a = idx & 511;
        size_t o = ((size_t)b * 4 + h) * 512 + a;
        float w = wt[o], wd = wtd[o];
        float jd = s_misc[h][0];
        float wdn = (1.f - jd) * wd + jd * w;
        wtd[o] = wdn;
        s_wtj[h][a] = s_misc[h][1] * w + s_misc[h][2] * wdn
                    + s_misc[h][3] * ((a == 0) ? 1.f : 0.f);
      }
      __syncthreads();
      if (tid < 128) {                            // normalize key
        int h = tid >> 5, m = tid & 31;
        s_kn[h][m] /= s_misc[h][7];
      }
      for (int a = tid; a < 512; a += BLK) {      // erase/add + column norms
        float n2 = 0.f;
        float wj0 = s_wtj[0][a], wj1 = s_wtj[1][a], wj2 = s_wtj[2][a], wj3 = s_wtj[3][a];
        for (int m = 0; m < 32; ++m) {
          size_t o = ((size_t)b * 32 + m) * 512 + a;
          float mv = mem[o];
          float prod = (1.f - s_er[0][m] * wj0) * (1.f - s_er[1][m] * wj1)
                     * (1.f - s_er[2][m] * wj2) * (1.f - s_er[3][m] * wj3);
          float ad = s_addv[0][m] * wj0 + s_addv[1][m] * wj1
                   + s_addv[2][m] * wj2 + s_addv[3][m] * wj3;
          mv = mv * prod + ad;
          mem[o] = mv;
          n2 += mv * mv;
        }
        s_norm2[a] = sqrtf(n2) + 1e-12f;
      }
      __syncthreads();
      for (int idx = tid; idx < 4 * 512; idx += BLK) {   // cosine similarity * beta
        int h = idx >> 9, a = idx & 511;
        float dot = 0.f;
        for (int m = 0; m < 32; ++m)
          dot += s_kn[h][m] * mem[((size_t)b * 32 + m) * 512 + a];
        s_z[h][a] = s_misc[h][5] * dot / s_norm2[a];
      }
      __syncthreads();
      if (tid < 4) {                              // softmax reduce per head
        float mx = -3.0e38f;
        for (int a = 0; a < 512; ++a) mx = fmaxf(mx, s_z[tid][a]);
        float sm = 0.f;
        for (int a = 0; a < 512; ++a) sm += expf(s_z[tid][a] - mx);
        s_red[tid] = mx; s_red[4 + tid] = sm;
      }
      __syncthreads();
      for (int idx = tid; idx < 4 * 512; idx += BLK) {   // wt_b, gate -> wt_c
        int h = idx >> 9, a = idx & 511;
        float wb = expf(s_z[h][a] - s_red[h]) / s_red[4 + h];
        float g = s_misc[h][6];
        s_z[h][a] = g * wb + (1.f - g) * s_wtj[h][a];
      }
      __syncthreads();
      for (int idx = tid; idx < 4 * 512; idx += BLK) {   // circ-conv + sharpen
        int h = idx >> 9, a = idx & 511;
        float wsv = s_shift[h][0] * s_z[h][(a + 1) & 511]
                  + s_shift[h][1] * s_z[h][a]
                  + s_shift[h][2] * s_z[h][(a + 511) & 511];
        s_wts[h][a] = powf(wsv + 1e-12f, s_misc[h][4]);
      }
      __syncthreads();
      if (tid < 4) {
        float sm = 0.f;
        for (int a = 0; a < 512; ++a) sm += s_wts[tid][a];
        s_red[tid] = sm;
      }
      __syncthreads();
      for (int idx = tid; idx < 4 * 512; idx += BLK) {   // renorm -> wt_new
        int h = idx >> 9, a = idx & 511;
        float wn = s_wts[h][a] / s_red[h];
        s_wts[h][a] = wn;
        wt[((size_t)b * 4 + h) * 512 + a] = wn;
      }
      __syncthreads();
      if (tid < 128) {                            // read for next step
        int h = tid >> 5, m = tid & 31;
        float dot = 0.f;
        for (int a = 0; a < 512; ++a)
          dot += s_wts[h][a] * mem[((size_t)b * 32 + m) * 512 + a];
        readv[b * 128 + h * 32 + m] = dot;
      }
      __syncthreads();
    }
    grid_barrier(barCnt, barGen);
  }
}

extern "C" void kernel_launch(void* const* d_in, const int* in_sizes, int n_in,
                              void* d_out, int out_size, void* d_ws, size_t ws_size,
                              hipStream_t stream) {
  const float* x  = (const float*)d_in[0];
  // d_in[1] = targets (unused by forward)
  const float* Ws = (const float*)d_in[2];
  const float* bs = (const float*)d_in[3];
  const float* Wo = (const float*)d_in[4];
  const float* bo = (const float*)d_in[5];
  const float* Wu = (const float*)d_in[6];
  const float* bu = (const float*)d_in[7];
  float* out = (float*)d_out;
  (void)in_sizes; (void)n_in; (void)out_size; (void)ws_size;

  char* ws = (char*)d_ws;
  size_t off = 0;
  auto carve = [&](size_t bytes) -> char* {
    char* p = ws + off;
    off = (off + bytes + 255) & ~(size_t)255;
    return p;
  };
  unsigned*       syncp = (unsigned*)      carve(256);
  unsigned short* Wpk   = (unsigned short*)carve((size_t)WPK_ELEMS * 2);
  unsigned short* comb  = (unsigned short*)carve((size_t)32 * KPAD * 2);
  float* state = (float*)carve((size_t)32 * 256 * 4);
  float* readv = (float*)carve((size_t)32 * 128 * 4);
  float* updg  = (float*)carve((size_t)32 * 424 * 4);
  float* wt    = (float*)carve((size_t)32 * 4 * 512 * 4);
  float* wtd   = (float*)carve((size_t)32 * 4 * 512 * 4);
  float* mem   = (float*)carve((size_t)32 * 32 * 512 * 4);

  dwm_pack_weights<<<(WPK_ELEMS + BLK - 1) / BLK, BLK, 0, stream>>>(Ws, Wo, Wu, Wpk);
  dwm_init<<<256, BLK, 0, stream>>>(state, readv, wt, wtd, mem, syncp);
  dwm_persistent<<<NWG, BLK, 0, stream>>>(x, bs, bo, bu, Wpk, comb,
                                          state, readv, updg, wt, wtd, mem,
                                          out, syncp);
}
